// UlyssesAttention_68006512165411
// MI455X (gfx1250) — matrix operand order repaired and checked by reading, not compiled, not run on hardware
//
#include <hip/hip_runtime.h>
#include <hip/hip_bf16.h>
#include <stdint.h>

// ---------------------------------------------------------------------------
// Types for CDNA5 WMMA (wave32): D(16x16 f32) = A(16x32 bf16) * B(32x16 bf16) + C
// ---------------------------------------------------------------------------
typedef __attribute__((ext_vector_type(16))) __bf16 v16bf;
typedef __attribute__((ext_vector_type(8)))  __bf16 v8bf;
typedef __attribute__((ext_vector_type(8)))  float  v8f;

static __device__ __forceinline__ v8f wmma_bf16(v16bf a, v16bf b, v8f c) {
    return __builtin_amdgcn_wmma_f32_16x16x32_bf16(
        /*neg_a=*/false, a, /*neg_b=*/false, b,
        /*c_mod=*/(short)0, c, /*reuse_a=*/false, /*reuse_b=*/false);
}

static __device__ __forceinline__ v16bf cat16(v8bf a, v8bf b) {
    v16bf r;
#pragma unroll
    for (int i = 0; i < 8; ++i) { r[i] = a[i]; r[i + 8] = b[i]; }
    return r;
}

static __device__ __forceinline__ v8f zero8() {
    v8f z;
#pragma unroll
    for (int i = 0; i < 8; ++i) z[i] = 0.0f;
    return z;
}

// Problem constants
#define DIMC      512
#define NHEADS    8
#define HDIM      64
#define TT        256
#define NN        256       // spatial tokens (16x16)
#define AA        16        // anchors
#define MM        (AA + NN) // 272 concat keys
#define MROWS     (TT * NN) // 65536
#define WINDOW    4

// ---------------------------------------------------------------------------
// Elementwise f32 -> bf16
// ---------------------------------------------------------------------------
__global__ __launch_bounds__(256) void cvt_bf16_kernel(const float* __restrict__ s,
                                                       __bf16* __restrict__ d, int n) {
    int i = blockIdx.x * 256 + threadIdx.x;
    if (i < n) d[i] = (__bf16)s[i];
}

// Scatter anchors into k (row-major, m<16) and v^T ([d][m], m<16)
__global__ __launch_bounds__(256) void anchor_scatter_kernel(const float* __restrict__ a,
                                                             __bf16* __restrict__ kx,
                                                             __bf16* __restrict__ vxT) {
    int i = blockIdx.x * 256 + threadIdx.x;         // [h][t][a][d] flat
    if (i >= NHEADS * TT * AA * HDIM) return;
    int d  = i & 63;
    int aa = (i >> 6) & 15;
    int t  = (i >> 10) & 255;
    int h  = i >> 18;
    __bf16 v = (__bf16)a[i];
    size_t ht = (size_t)(h * TT + t);
    kx [(ht * MM + aa) * HDIM + d] = v;
    vxT[(ht * HDIM + d) * MM + aa] = v;
}

// ---------------------------------------------------------------------------
// Tiled bf16 GEMM:  C[M x Nout] = A[M x 512] * B[512 x Nout] + bias
//   A_F32: A is fp32 in global, converted to bf16 on the way into LDS
//   QKV:   epilogue scatters into q/k/v^T head-major bf16 buffers
//   else:  epilogue writes fp32 to Cout (the final output)
// Block: 256 threads = 8 waves (4 M x 2 N); tile 128x128, K-step 32.
// Double-buffered LDS: one barrier per K-step, next slice prefetched into
// registers before the WMMAs of the current slice.
// ---------------------------------------------------------------------------
template <bool A_F32, bool QKV>
__global__ __launch_bounds__(256) void gemm_bf16_kernel(
    const void* __restrict__ Aptr, const __bf16* __restrict__ B,
    const float* __restrict__ bias, float* __restrict__ Cout,
    __bf16* __restrict__ qb, __bf16* __restrict__ kb, __bf16* __restrict__ vb,
    int Nout) {
    constexpr int K = DIMC;
    constexpr int NSTEP = K / 32;                     // 16
    __shared__ __align__(16) __bf16 As [2][128][40];  // [buf][m][k], 80B rows
    __shared__ __align__(16) __bf16 Bst[2][128][40];  // [buf][n][k] transposed

    const int tid  = threadIdx.x;
    const int lane = tid & 31;
    const int w    = tid >> 5;
    const int lr   = lane & 15;
    const int hi   = lane >> 4;
    const int wm   = w & 3;              // wave M index (0..3) -> 32 rows
    const int wn   = w >> 2;             // wave N index (0..1) -> 64 cols
    const int mWG  = blockIdx.y * 128;
    const int nWG  = blockIdx.x * 128;

    const int arow = tid >> 1;           // 0..127
    const int acol = (tid & 1) * 16;     // 0 or 16
    const int brow = tid >> 3;           // 0..31  (k)
    const int bcol = (tid & 7) * 16;     // 0..112 (n)

    v8f acc[2][4];
#pragma unroll
    for (int i = 0; i < 2; ++i)
#pragma unroll
        for (int j = 0; j < 4; ++j) acc[i][j] = zero8();

    // prefetch registers
    float4 pf0, pf1, pf2, pf3;
    v8bf   pa0, pa1, pb0, pb1;

    auto gload = [&](int k0) {
        if (A_F32) {
            const float* ag = (const float*)Aptr + (size_t)(mWG + arow) * K + k0 + acol;
            const float4* ag4 = (const float4*)ag;
            pf0 = ag4[0]; pf1 = ag4[1]; pf2 = ag4[2]; pf3 = ag4[3];
        } else {
            const __bf16* ag = (const __bf16*)Aptr + (size_t)(mWG + arow) * K + k0 + acol;
            pa0 = *(const v8bf*)ag;
            pa1 = *(const v8bf*)(ag + 8);
        }
        const __bf16* bg = B + (size_t)(k0 + brow) * Nout + nWG + bcol;
        pb0 = *(const v8bf*)bg;
        pb1 = *(const v8bf*)(bg + 8);
    };
    auto lstore = [&](int buf) {
        if (A_F32) {
            __bf16* dst = &As[buf][arow][acol];
            dst[0]=(__bf16)pf0.x; dst[1]=(__bf16)pf0.y; dst[2]=(__bf16)pf0.z; dst[3]=(__bf16)pf0.w;
            dst[4]=(__bf16)pf1.x; dst[5]=(__bf16)pf1.y; dst[6]=(__bf16)pf1.z; dst[7]=(__bf16)pf1.w;
            dst[8]=(__bf16)pf2.x; dst[9]=(__bf16)pf2.y; dst[10]=(__bf16)pf2.z; dst[11]=(__bf16)pf2.w;
            dst[12]=(__bf16)pf3.x; dst[13]=(__bf16)pf3.y; dst[14]=(__bf16)pf3.z; dst[15]=(__bf16)pf3.w;
        } else {
            *(v8bf*)&As[buf][arow][acol]     = pa0;
            *(v8bf*)&As[buf][arow][acol + 8] = pa1;
        }
#pragma unroll
        for (int i2 = 0; i2 < 8; ++i2) {
            Bst[buf][bcol + i2][brow]     = pb0[i2];
            Bst[buf][bcol + 8 + i2][brow] = pb1[i2];
        }
    };

    // prologue: slice 0 into buffer 0
    gload(0);
    lstore(0);
    __syncthreads();

    for (int step = 0; step < NSTEP; ++step) {
        const int cur = step & 1;
        // prefetch next slice into registers (overlaps with WMMAs below)
        if (step < NSTEP - 1) gload((step + 1) * 32);

        // ---- fragments + 8 WMMAs from current buffer ----
        v16bf af[2], bfr[4];
#pragma unroll
        for (int i = 0; i < 2; ++i) {
            const __bf16* pa_ = &As[cur][wm * 32 + i * 16 + lr][0];
            af[i] = cat16(*(const v8bf*)(pa_ + 8 * hi),
                          *(const v8bf*)(pa_ + 16 + 8 * hi));
        }
#pragma unroll
        for (int j = 0; j < 4; ++j) {
            const __bf16* pb_ = &Bst[cur][wn * 64 + j * 16 + lr][0];
            bfr[j] = cat16(*(const v8bf*)(pb_ + 8 * hi),
                           *(const v8bf*)(pb_ + 16 + 8 * hi));
        }
#pragma unroll
        for (int i = 0; i < 2; ++i)
#pragma unroll
            for (int j = 0; j < 4; ++j)
                acc[i][j] = wmma_bf16(af[i], bfr[j], acc[i][j]);

        // stage prefetched slice into the other buffer
        if (step < NSTEP - 1) lstore(cur ^ 1);
        __syncthreads();
    }

    // ---- epilogue ----
#pragma unroll
    for (int i = 0; i < 2; ++i) {
#pragma unroll
        for (int j = 0; j < 4; ++j) {
            const int col = nWG + wn * 64 + j * 16 + lr;
            const float bv = bias[col];
            if (QKV) {
                const int s    = col >> 9;
                const int cc   = col & 511;
                const int head = cc >> 6;
                const int d    = cc & 63;
#pragma unroll
                for (int r = 0; r < 8; ++r) {
                    const int row = mWG + wm * 32 + i * 16 + r + 8 * hi;
                    const int t = row >> 8, n = row & 255;
                    const float val = acc[i][j][r] + bv;
                    const size_t ht = (size_t)(head * TT + t);
                    if (s == 0)      qb[(ht * NN + n) * HDIM + d]        = (__bf16)val;
                    else if (s == 1) kb[(ht * MM + AA + n) * HDIM + d]   = (__bf16)val;
                    else             vb[(ht * HDIM + d) * MM + AA + n]   = (__bf16)val;
                }
            } else {
#pragma unroll
                for (int r = 0; r < 8; ++r) {
                    const int row = mWG + wm * 32 + i * 16 + r + 8 * hi;
                    Cout[(size_t)row * Nout + col] = acc[i][j][r] + bv;
                }
            }
        }
    }
}

// ---------------------------------------------------------------------------
// Sparse windowed attention. One wave per (head, t, 16-row query tile).
// Key set per tile: anchor tile + 3 spatial tiles (window +-4 fits in them).
// QK^T: 8 WMMAs; masked softmax via 16-lane shuffles; PV: 8 WMMAs.
// ---------------------------------------------------------------------------
__global__ __launch_bounds__(256) void attn_kernel(const __bf16* __restrict__ q,
                                                   const __bf16* __restrict__ kx,
                                                   const __bf16* __restrict__ vxT,
                                                   __bf16* __restrict__ o) {
    __shared__ __align__(16) __bf16 pls[8][16][72];   // per-wave P tile, 144B rows

    const int tid  = threadIdx.x;
    const int lane = tid & 31;
    const int w    = tid >> 5;
    const int lr   = lane & 15;
    const int hi   = lane >> 4;

    const int gw = blockIdx.x * 8 + w;     // 32768 waves total
    const int h  = gw >> 12;
    const int t  = (gw >> 4) & 255;
    const int nt = gw & 15;
    const int n0 = nt * 16;
    const size_t ht = (size_t)(h * TT + t);

    // --- Q fragments (A: 16 rows x K=64 -> two K=32 chunks) ---
    const __bf16* qrow = q + (ht * NN + n0 + lr) * HDIM;
    v16bf qa[2];
#pragma unroll
    for (int c = 0; c < 2; ++c)
        qa[c] = cat16(*(const v8bf*)(qrow + c * 32 + 8 * hi),
                      *(const v8bf*)(qrow + c * 32 + 16 + 8 * hi));

    // --- key tile map: 0 = anchors, 1..3 = spatial tiles nt-1..nt+1 ---
    int  mreal[4];
    bool valid[4];
    int  jb[4];
    mreal[0] = 0; valid[0] = true; jb[0] = 0;
#pragma unroll
    for (int kk = 1; kk < 4; ++kk) {
        const int jt = nt + kk - 2;
        valid[kk] = (jt >= 0) && (jt < 16);
        const int jc = jt < 0 ? 0 : (jt > 15 ? 15 : jt);
        mreal[kk] = AA + jc * 16;
        jb[kk]    = jt * 16;
    }

    // --- scores: 16 x 64 (4 accumulator tiles) ---
    v8f S[4];
#pragma unroll
    for (int kk = 0; kk < 4; ++kk) {
        S[kk] = zero8();
        const __bf16* krow = kx + (ht * MM + mreal[kk] + lr) * HDIM;
#pragma unroll
        for (int c = 0; c < 2; ++c) {
            v16bf b = cat16(*(const v8bf*)(krow + c * 32 + 8 * hi),
                            *(const v8bf*)(krow + c * 32 + 16 + 8 * hi));
            S[kk] = wmma_bf16(qa[c], b, S[kk]);
        }
    }

    // --- scale + mask ---
    const float scale = 0.125f;   // 1/sqrt(64)
#pragma unroll
    for (int kk = 0; kk < 4; ++kk) {
#pragma unroll
        for (int r = 0; r < 8; ++r) {
            float s = S[kk][r] * scale;
            if (kk > 0) {
                const int n  = n0 + r + 8 * hi;
                const int j  = jb[kk] + lr;
                const int df = n - j;
                const bool ok = valid[kk] && (df >= -WINDOW) && (df <= WINDOW);
                if (!ok) s = -1e30f;
            }
            S[kk][r] = s;
        }
    }

    // --- softmax over 64 cols (row lives on 16 lanes of a half-wave) ---
#pragma unroll
    for (int r = 0; r < 8; ++r) {
        float a0 = S[0][r], a1 = S[1][r], a2 = S[2][r], a3 = S[3][r];
        float mx = fmaxf(fmaxf(a0, a1), fmaxf(a2, a3));
#pragma unroll
        for (int off = 1; off < 16; off <<= 1)
            mx = fmaxf(mx, __shfl_xor(mx, off, 16));
        const float e0 = __expf(a0 - mx), e1 = __expf(a1 - mx);
        const float e2 = __expf(a2 - mx), e3 = __expf(a3 - mx);
        float sm = (e0 + e1) + (e2 + e3);
#pragma unroll
        for (int off = 1; off < 16; off <<= 1)
            sm += __shfl_xor(sm, off, 16);
        const float inv = 1.0f / sm;
        const int rr = r + 8 * hi;
        pls[w][rr][lr]      = (__bf16)(e0 * inv);
        pls[w][rr][16 + lr] = (__bf16)(e1 * inv);
        pls[w][rr][32 + lr] = (__bf16)(e2 * inv);
        pls[w][rr][48 + lr] = (__bf16)(e3 * inv);
    }

    // --- P fragments from LDS (A layout), then PV with v^T B-fragments ---
    v16bf pa[2];
#pragma unroll
    for (int c = 0; c < 2; ++c) {
        const __bf16* pr = &pls[w][lr][0];
        pa[c] = cat16(*(const v8bf*)(pr + c * 32 + 16 * hi),
                      *(const v8bf*)(pr + c * 32 + 16 * hi + 8));
    }

    const __bf16* vbase = vxT + ht * HDIM * MM;
    v8f O[4];
#pragma unroll
    for (int dt = 0; dt < 4; ++dt) {
        O[dt] = zero8();
        const __bf16* vrow = vbase + (size_t)(dt * 16 + lr) * MM;
#pragma unroll
        for (int c = 0; c < 2; ++c) {
            const int mb = mreal[2 * c + hi];   // packed K half -> real key tile
            v16bf b = cat16(*(const v8bf*)(vrow + mb),
                            *(const v8bf*)(vrow + mb + 8));
            O[dt] = wmma_bf16(pa[c], b, O[dt]);
        }
    }

    // --- store o in [t*256+n][512] layout (bf16) for the proj GEMM ---
#pragma unroll
    for (int dt = 0; dt < 4; ++dt) {
#pragma unroll
        for (int r = 0; r < 8; ++r) {
            const int n = n0 + r + 8 * hi;
            o[((size_t)(t * NN + n)) * DIMC + h * HDIM + dt * 16 + lr] = (__bf16)O[dt][r];
        }
    }
}

// ---------------------------------------------------------------------------
extern "C" void kernel_launch(void* const* d_in, const int* in_sizes, int n_in,
                              void* d_out, int out_size, void* d_ws, size_t ws_size,
                              hipStream_t stream) {
    const float* x       = (const float*)d_in[0];
    const float* anchors = (const float*)d_in[1];
    const float* w_qkv   = (const float*)d_in[2];
    const float* b_qkv   = (const float*)d_in[3];
    const float* w_proj  = (const float*)d_in[4];
    const float* b_proj  = (const float*)d_in[5];
    float* out = (float*)d_out;

    char* p = (char*)d_ws;
    auto take = [&](size_t bytes) -> char* {
        char* r = p;
        p += (bytes + 255) & ~(size_t)255;
        return r;
    };
    __bf16* wqkv_bf  = (__bf16*)take((size_t)DIMC * 3 * DIMC * 2);
    __bf16* wproj_bf = (__bf16*)take((size_t)DIMC * DIMC * 2);
    __bf16* q_bf     = (__bf16*)take((size_t)NHEADS * TT * NN * HDIM * 2);
    __bf16* kx_bf    = (__bf16*)take((size_t)NHEADS * TT * MM * HDIM * 2);
    __bf16* vxT_bf   = (__bf16*)take((size_t)NHEADS * TT * HDIM * MM * 2);
    __bf16* o_bf     = (__bf16*)take((size_t)MROWS * DIMC * 2);

    // weight / anchor conversion
    {
        int n1 = DIMC * 3 * DIMC;
        cvt_bf16_kernel<<<(n1 + 255) / 256, 256, 0, stream>>>(w_qkv, wqkv_bf, n1);
        int n2 = DIMC * DIMC;
        cvt_bf16_kernel<<<(n2 + 255) / 256, 256, 0, stream>>>(w_proj, wproj_bf, n2);
        int n3 = NHEADS * TT * AA * HDIM;
        anchor_scatter_kernel<<<(n3 + 255) / 256, 256, 0, stream>>>(anchors, kx_bf, vxT_bf);
    }

    // QKV GEMM: x(f32) -> q/k/v^T (bf16, head-major, anchor-concat slots)
    {
        dim3 g(3 * DIMC / 128, MROWS / 128);
        gemm_bf16_kernel<true, true><<<g, 256, 0, stream>>>(
            (const void*)x, wqkv_bf, b_qkv, nullptr, q_bf, kx_bf, vxT_bf, 3 * DIMC);
    }

    // sparse windowed attention
    {
        int nwaves = NHEADS * TT * 16;  // 32768
        attn_kernel<<<nwaves / 8, 256, 0, stream>>>(q_bf, kx_bf, vxT_bf, o_bf);
    }

    // projection GEMM -> fp32 output
    {
        dim3 g(DIMC / 128, MROWS / 128);
        gemm_bf16_kernel<false, false><<<g, 256, 0, stream>>>(
            (const void*)o_bf, wproj_bf, b_proj, out, nullptr, nullptr, nullptr, DIMC);
    }
}
